// GRUDecoderAtt_52381421142201
// MI455X (gfx1250) — matrix operand order repaired and checked by reading, not compiled, not run on hardware
//
#include <hip/hip_runtime.h>

#define DEV __device__ __forceinline__

typedef __attribute__((ext_vector_type(16))) __bf16 v16bf;
typedef __attribute__((ext_vector_type(8)))  __bf16 v8bf;
typedef __attribute__((ext_vector_type(8)))  float  v8f;

constexpr int kB = 32, kT = 32, kV = 32000, kE = 512, kH = 512, kS = 512;
constexpr int kC = 256, kVF = 512, kA = 256, kL = 50;
constexpr int kIN = kE + kS + kC + kVF;   // 1792
constexpr int kKG = kIN + kH;             // 2304

// ---------------- workspace layout (bytes, all 256B aligned) ----------------
constexpr size_t OFF_BAR   = 0;                                   // 128 * 4
constexpr size_t OFF_SP    = 512;                                 // f32 B*L*A
constexpr size_t OFF_XEMB  = OFF_SP    + (size_t)kB*kL*kA*4;      // bf16 T*B*E
constexpr size_t OFF_ASTEP = OFF_XEMB  + (size_t)kT*kB*kE*2;      // bf16 B*2304
constexpr size_t OFF_H2ALL = OFF_ASTEP + (size_t)kB*kKG*2;        // f32 T*B*H
constexpr size_t OFF_H2BF  = OFF_H2ALL + (size_t)kT*kB*kH*4;      // bf16 T*B*H
constexpr size_t OFF_SOBF  = OFF_H2BF  + (size_t)kT*kB*kH*2;      // bf16 B*L*S
constexpr size_t OFF_WST   = OFF_SOBF  + (size_t)kB*kL*kS*2;      // bf16 A x S (Ws^T)
constexpr size_t OFF_WHT   = OFF_WST   + (size_t)kA*kS*2;         // bf16 A x H (Wh^T)
constexpr size_t OFF_DECBF = OFF_WHT   + (size_t)kA*kH*2;         // bf16 B*H
constexpr size_t OFF_QBUF  = OFF_DECBF + (size_t)kB*kH*2;         // f32 B*A
constexpr size_t OFF_WG    = OFF_QBUF  + (size_t)kB*kA*4;         // bf16 1536*2304
constexpr size_t OFF_WOUT  = OFF_WG    + (size_t)3*kH*kKG*2;      // bf16 V*H
constexpr size_t NEED_WG   = OFF_WOUT;
constexpr size_t NEED_FULL = OFF_WOUT  + (size_t)kV*kH*2;

// ---------------- helpers ----------------
DEV __bf16 f2bf(float f) {
  unsigned u = __float_as_uint(f);
  unsigned r = (u + 0x7FFFu + ((u >> 16) & 1u)) >> 16;
  unsigned short s = (unsigned short)r;
  return __builtin_bit_cast(__bf16, s);
}

DEV v8bf pack8(const float* p) {   // 8 contiguous f32 -> 8 bf16
  float4 f0 = *(const float4*)p, f1 = *(const float4*)(p + 4);
  v8bf r;
  r[0]=f2bf(f0.x); r[1]=f2bf(f0.y); r[2]=f2bf(f0.z); r[3]=f2bf(f0.w);
  r[4]=f2bf(f1.x); r[5]=f2bf(f1.y); r[6]=f2bf(f1.z); r[7]=f2bf(f1.w);
  return r;
}

DEV v8f vzero8() {
  v8f v;
#pragma unroll
  for (int i = 0; i < 8; ++i) v[i] = 0.0f;
  return v;
}

// A-fragment: 16x32 bf16 tile from row-major [M x ld] bf16 matrix.
DEV v16bf load_afrag(const __bf16* A, int mbase, int ld, int kbase, int lane) {
  int row  = mbase + (lane & 15);
  int koff = kbase + ((lane >> 4) << 4);
  const __bf16* p = A + (size_t)row * ld + koff;
  v8bf lo = *(const v8bf*)p;
  v8bf hi = *(const v8bf*)(p + 8);
  return __builtin_shufflevector(lo, hi, 0,1,2,3,4,5,6,7,8,9,10,11,12,13,14,15);
}

DEV v16bf load_bfrag_bf16(const __bf16* p) { return *(const v16bf*)p; }

DEV v16bf load_bfrag_f32(const float* p) {   // fallback on-the-fly conversion
  v16bf r;
#pragma unroll
  for (int i = 0; i < 16; i += 4) {
    float4 f = *(const float4*)(p + i);
    r[i] = f2bf(f.x); r[i+1] = f2bf(f.y); r[i+2] = f2bf(f.z); r[i+3] = f2bf(f.w);
  }
  return r;
}

DEV v8f wmma_bf16(v16bf a, v16bf b, v8f c) {
  return __builtin_amdgcn_wmma_f32_16x16x32_bf16(false, a, false, b, (short)0, c, false, false);
}

DEV void gridbar(unsigned* bar, int slot, unsigned nwg) {
  __syncthreads();
  if (threadIdx.x == 0) {
    __threadfence();
    atomicAdd(&bar[slot], 1u);
    volatile unsigned* vb = bar + slot;
    while (*vb < nwg) { __builtin_amdgcn_s_sleep(1); }
  }
  __syncthreads();
  __threadfence();
}

// ---------------- prep kernels ----------------
__global__ __launch_bounds__(128) void gd_init(unsigned* bar) {
  bar[threadIdx.x] = 0u;
}

// transpose+convert Ws->WsT, Wh->WhT (bf16 [A][S]); convert decoder_state->bf16
__global__ __launch_bounds__(64) void gd_cvt_small(const float* __restrict__ Ws,
                                                   const float* __restrict__ Wh,
                                                   const float* __restrict__ dec,
                                                   unsigned short* __restrict__ wst_u,
                                                   unsigned short* __restrict__ wht_u,
                                                   unsigned short* __restrict__ decbf_u) {
  int idx = blockIdx.x, tid = threadIdx.x;
  if (idx < 512) {
    const float* src = (idx < 256) ? Ws : Wh;
    __bf16* dst = (__bf16*)((idx < 256) ? wst_u : wht_u) + (size_t)(idx & 255) * kS;
    int a = idx & 255, s0 = tid * 8;
    v8bf r;
#pragma unroll
    for (int j = 0; j < 8; ++j) r[j] = f2bf(src[(size_t)(s0 + j) * kA + a]);
    *(v8bf*)(dst + s0) = r;
  } else {
    int b = idx - 512;
    const float* src = dec + (size_t)b * kH;
    __bf16* dst = (__bf16*)decbf_u + (size_t)b * kH;
    int s0 = tid * 8;
    *(v8bf*)(dst + s0) = pack8(src + s0);
  }
}

// session_outputs -> bf16
__global__ __launch_bounds__(64) void gd_socvt(const float* __restrict__ so,
                                               unsigned short* __restrict__ sobf_u) {
  __bf16* dst = (__bf16*)sobf_u + (size_t)blockIdx.x * kS;
  const float* src = so + (size_t)blockIdx.x * kS;
  int s0 = threadIdx.x * 8;
  *(v8bf*)(dst + s0) = pack8(src + s0);
}

// Wg[n][k] bf16, k<1792 from W_ih, else W_hh
__global__ __launch_bounds__(256) void gd_wgcvt(const float* __restrict__ Wih,
                                                const float* __restrict__ Whh,
                                                unsigned short* __restrict__ wg_u) {
  int n = blockIdx.x;
  __bf16* dst = (__bf16*)wg_u + (size_t)n * kKG;
  const float* sih = Wih + (size_t)n * kIN;
  const float* shh = Whh + (size_t)n * kH;
  for (int c = threadIdx.x; c < kKG / 8; c += 256) {
    int k0 = c * 8;
    const float* src = (k0 < kIN) ? sih + k0 : shh + (k0 - kIN);
    *(v8bf*)(dst + k0) = pack8(src);
  }
}

// W_out f32 (V x H) -> bf16
__global__ __launch_bounds__(256) void gd_woutcvt(const float* __restrict__ Wout,
                                                  unsigned short* __restrict__ wo_u) {
  size_t base = ((size_t)blockIdx.x * 256 + threadIdx.x) * 8;
  *(v8bf*)((__bf16*)wo_u + base) = pack8(Wout + base);
}

// xemb[t][b][e] = bf16(emb[tgt[b][t]][e])
__global__ __launch_bounds__(64) void gd_xemb(const int* __restrict__ tgt,
                                              const float* __restrict__ emb,
                                              unsigned short* __restrict__ xemb_u) {
  int idx = blockIdx.x;           // t*kB + b
  int b = idx & 31, t = idx >> 5;
  int tok = tgt[b * kT + t];
  const float* src = emb + (size_t)tok * kE;
  __bf16* dst = (__bf16*)xemb_u + (size_t)idx * kE;
  int s0 = threadIdx.x * 8;
  *(v8bf*)(dst + s0) = pack8(src + s0);
}

// sess_proj = session_outputs @ Ws + b : M=1600, N=256, K=512 WMMA GEMM
__global__ __launch_bounds__(256) void gd_sessproj(const unsigned short* __restrict__ sobf_u,
                                                   const unsigned short* __restrict__ wst_u,
                                                   const float* __restrict__ ab,
                                                   float* __restrict__ sp) {
  const __bf16* A  = (const __bf16*)sobf_u;
  const __bf16* Bt = (const __bf16*)wst_u;
  const int wv = threadIdx.x >> 5, ln = threadIdx.x & 31;
  const int job = blockIdx.x * 8 + wv;     // 400 jobs
  const int mbase = (job >> 2) * 16;       // 100 M-tiles
  const int nbase = (job & 3) * 64;        // 4 chunks of 64
  const int nl = ln & 15, kl = (ln >> 4) << 4;
  v8f a0 = vzero8(), a1 = vzero8(), a2 = vzero8(), a3 = vzero8();
  v16bf af = load_afrag(A, mbase, kS, 0, ln);
  v16bf b0 = load_bfrag_bf16(Bt + (size_t)(nbase +  0 + nl) * kS + kl);
  v16bf b1 = load_bfrag_bf16(Bt + (size_t)(nbase + 16 + nl) * kS + kl);
  v16bf b2 = load_bfrag_bf16(Bt + (size_t)(nbase + 32 + nl) * kS + kl);
  v16bf b3 = load_bfrag_bf16(Bt + (size_t)(nbase + 48 + nl) * kS + kl);
  for (int ks = 0; ks < kS / 32 - 1; ++ks) {
    int kc = (ks + 1) * 32 + kl;
    v16bf afn = load_afrag(A, mbase, kS, (ks + 1) * 32, ln);
    v16bf b0n = load_bfrag_bf16(Bt + (size_t)(nbase +  0 + nl) * kS + kc);
    v16bf b1n = load_bfrag_bf16(Bt + (size_t)(nbase + 16 + nl) * kS + kc);
    v16bf b2n = load_bfrag_bf16(Bt + (size_t)(nbase + 32 + nl) * kS + kc);
    v16bf b3n = load_bfrag_bf16(Bt + (size_t)(nbase + 48 + nl) * kS + kc);
    a0 = wmma_bf16(af, b0, a0); a1 = wmma_bf16(af, b1, a1);
    a2 = wmma_bf16(af, b2, a2); a3 = wmma_bf16(af, b3, a3);
    af = afn; b0 = b0n; b1 = b1n; b2 = b2n; b3 = b3n;
  }
  a0 = wmma_bf16(af, b0, a0); a1 = wmma_bf16(af, b1, a1);
  a2 = wmma_bf16(af, b2, a2); a3 = wmma_bf16(af, b3, a3);
  v8f accs[4] = {a0, a1, a2, a3};
#pragma unroll
  for (int q = 0; q < 4; ++q) {
    int n = nbase + q * 16 + nl;
    float bo = ab[n];
#pragma unroll
    for (int i = 0; i < 8; ++i) {
      int row = mbase + i + 8 * (ln >> 4);
      sp[(size_t)row * kA + n] = accs[q][i] + bo;
    }
  }
}

// ---------------- persistent recurrence ----------------
template <bool PRE>
__global__ __launch_bounds__(256) void gd_recur(
    const float* __restrict__ dec, const float* __restrict__ cat,
    const float* __restrict__ vfeat, const float* __restrict__ so,
    const unsigned char* __restrict__ mask, const float* __restrict__ av,
    const float* __restrict__ Wih, const float* __restrict__ Whh,
    const float* __restrict__ bih, const float* __restrict__ bhh,
    const float* __restrict__ sessproj, const unsigned short* __restrict__ xemb_u,
    unsigned short* __restrict__ astep_u, const unsigned short* __restrict__ wg_u,
    const unsigned short* __restrict__ wht_u, const unsigned short* __restrict__ decbf_u,
    float* __restrict__ qbuf, float* __restrict__ h2all,
    unsigned short* __restrict__ h2bf_u, float* __restrict__ outH,
    unsigned* __restrict__ bar) {
  const __bf16* xe    = (const __bf16*)xemb_u;
  const __bf16* Wg    = (const __bf16*)wg_u;
  const __bf16* WhT   = (const __bf16*)wht_u;
  const __bf16* decbf = (const __bf16*)decbf_u;
  __bf16* Astep       = (__bf16*)astep_u;
  __bf16* h2bf        = (__bf16*)h2bf_u;

  __shared__ float sh_h[kH];
  __shared__ float sh_q[kA];
  __shared__ float sh_e[64];
  __shared__ float sh_ctx[kS];
  __shared__ float sh_g[4][32][16];

  const int tid = threadIdx.x;
  const int wg  = blockIdx.x;          // 0..31
  const int wv  = tid >> 5, ln = tid & 31;
  const int nl  = ln & 15, kl = (ln >> 4) << 4;

  for (int t = 0; t < kT; ++t) {
    // ---- Phase A0: q = h_prev @ Wh via WMMA; WG w -> tile (mt=w&1, nt=w>>1), wave 0
    if (wv == 0) {
      const __bf16* Ab = (t == 0) ? decbf : h2bf + (size_t)(t - 1) * kB * kH;
      const int mt = wg & 1, nt = wg >> 1;
      const int mb = mt * 16;
      const __bf16* Bb = WhT + (size_t)(nt * 16 + nl) * kH + kl;
      v8f qa = vzero8();
      v16bf af = load_afrag(Ab, mb, kH, 0, ln);
      v16bf bf_ = load_bfrag_bf16(Bb);
      for (int ks = 0; ks < kH / 32 - 1; ++ks) {
        v16bf afn = load_afrag(Ab, mb, kH, (ks + 1) * 32, ln);
        v16bf bfn = load_bfrag_bf16(Bb + (ks + 1) * 32);
        qa = wmma_bf16(af, bf_, qa);
        af = afn; bf_ = bfn;
      }
      qa = wmma_bf16(af, bf_, qa);
#pragma unroll
      for (int i = 0; i < 8; ++i)
        qbuf[(size_t)(mb + i + 8 * (ln >> 4)) * kA + nt * 16 + nl] = qa[i];
    }
    gridbar(bar, 3 * t, kB);

    // ---- Phase A1: attention + build A-matrix row (WG = batch b) ----
    const int b = wg;
    const float* hprev = (t == 0) ? dec + (size_t)b * kH
                                  : h2all + ((size_t)(t - 1) * kB + b) * kH;
    for (int i = tid; i < kH; i += 256) sh_h[i] = hprev[i];
    sh_q[tid] = qbuf[(size_t)b * kA + tid];
    __syncthreads();
    for (int l = wv; l < kL; l += 8) {
      const float* sp = sessproj + ((size_t)b * kL + l) * kA;
      float s = 0.0f;
      for (int a = ln; a < kA; a += 32) s += av[a] * tanhf(sh_q[a] + sp[a]);
#pragma unroll
      for (int off = 16; off > 0; off >>= 1) s += __shfl_xor(s, off, 32);
      if (ln == 0) sh_e[l] = mask[b * kL + l] ? s : -3.0e38f;
    }
    __syncthreads();
    if (tid == 0) {
      float mx = -3.4e38f;
      for (int l = 0; l < kL; ++l) mx = fmaxf(mx, sh_e[l]);
      float sum = 0.0f;
      for (int l = 0; l < kL; ++l) { float ex = expf(sh_e[l] - mx); sh_e[l] = ex; sum += ex; }
      float inv = 1.0f / sum;
      for (int l = 0; l < kL; ++l) sh_e[l] *= inv;
    }
    __syncthreads();
    for (int s = tid; s < kS; s += 256) {
      float c = 0.0f;
      for (int l = 0; l < kL; ++l) c += sh_e[l] * so[((size_t)b * kL + l) * kS + s];
      sh_ctx[s] = c;
    }
    __syncthreads();
    {
      __bf16* Arow = Astep + (size_t)b * kKG;
      const __bf16* xrow = xe + ((size_t)t * kB + b) * kE;
      for (int i = tid; i < kE;  i += 256) Arow[i]        = xrow[i];
      for (int i = tid; i < kS;  i += 256) Arow[kE + i]   = f2bf(sh_ctx[i]);
      for (int i = tid; i < kC;  i += 256) Arow[1024 + i] = f2bf(cat[(size_t)b * kC + i]);
      for (int i = tid; i < kVF; i += 256) Arow[1280 + i] = f2bf(vfeat[(size_t)b * kVF + i]);
      for (int i = tid; i < kH;  i += 256) Arow[kIN + i]  = f2bf(sh_h[i]);
    }
    gridbar(bar, 3 * t + 1, kB);

    // ---- Phase B: gate GEMMs (WG = 16 cols; 8 waves = 4 gates x 2 M-tiles) ----
    const int colb = wg * 16;
    const int g  = wv >> 1;            // 0=r 1=z 2=xn 3=hn
    const int mbase = (wv & 1) * 16;
    const int nbaseW = (g >= 2 ? 2 : g) * kH + colb;
    const int ks0 = (g == 3) ? kIN / 32 : 0;
    const int ks1 = (g == 2) ? kIN / 32 : kKG / 32;
    const int n = nbaseW + nl;
    auto loadB = [&](int ks) -> v16bf {
      int kc = ks * 32 + kl;
      if (PRE) return load_bfrag_bf16(Wg + (size_t)n * kKG + kc);
      const float* brow = (kc < kIN) ? Wih + (size_t)n * kIN + kc
                                     : Whh + (size_t)n * kH + (kc - kIN);
      return load_bfrag_f32(brow);
    };
    v8f acc = vzero8();
    v16bf af = load_afrag(Astep, mbase, kKG, ks0 * 32, ln);
    v16bf bf_ = loadB(ks0);
    for (int ks = ks0; ks < ks1 - 1; ++ks) {
      v16bf afn = load_afrag(Astep, mbase, kKG, (ks + 1) * 32, ln);
      v16bf bfn = loadB(ks + 1);
      acc = wmma_bf16(af, bf_, acc);
      af = afn; bf_ = bfn;
    }
    acc = wmma_bf16(af, bf_, acc);
#pragma unroll
    for (int i = 0; i < 8; ++i)
      sh_g[g][mbase + i + 8 * (ln >> 4)][nl] = acc[i];
    __syncthreads();
    // combine gates -> h2
    for (int e = tid; e < 512; e += 256) {
      int m = e >> 4, j = e & 15, col = colb + j;
      const float* hp = (t == 0) ? dec + (size_t)m * kH
                                 : h2all + ((size_t)(t - 1) * kB + m) * kH;
      float hv = hp[col];
      float gr = sh_g[0][m][j] + bih[col] + bhh[col];
      float gz = sh_g[1][m][j] + bih[kH + col] + bhh[kH + col];
      float xn = sh_g[2][m][j] + bih[2 * kH + col];
      float hn = sh_g[3][m][j] + bhh[2 * kH + col];
      float r = 1.0f / (1.0f + expf(-gr));
      float z = 1.0f / (1.0f + expf(-gz));
      float nn = tanhf(xn + r * hn);
      float h2 = (1.0f - z) * nn + z * hv;
      size_t ridx = ((size_t)t * kB + m) * kH + col;
      h2all[ridx] = h2;
      h2bf[ridx]  = f2bf(h2);
      outH[((size_t)m * kT + t) * kH + col] = h2;
    }
    gridbar(bar, 3 * t + 2, kB);
  }
}

// ---------------- vocab projection: 1024 x 32000 x 512, 32x64 tile per wave ---
template <bool PRE>
__global__ __launch_bounds__(256) void gd_vocab(const unsigned short* __restrict__ h2bf_u,
                                                const unsigned short* __restrict__ wo_u,
                                                const float* __restrict__ Wout,
                                                const float* __restrict__ bout,
                                                float* __restrict__ out) {
  const __bf16* A  = (const __bf16*)h2bf_u;
  const __bf16* Wo = (const __bf16*)wo_u;
  const int wv = threadIdx.x >> 5, ln = threadIdx.x & 31;
  const int job = blockIdx.x * 8 + wv;   // 0..15999
  const int mbase = (job / 500) * 32;    // 32 M-tiles of 32 rows
  const int nbase = (job % 500) * 64;    // 500 chunks of 64 cols
  const int nl = ln & 15, kl = (ln >> 4) << 4;
  auto loadB = [&](int q, int ks) -> v16bf {
    size_t off = (size_t)(nbase + q * 16 + nl) * kH + ks * 32 + kl;
    if (PRE) return load_bfrag_bf16(Wo + off);
    return load_bfrag_f32(Wout + off);
  };
  v8f acc[2][4];
#pragma unroll
  for (int mi = 0; mi < 2; ++mi)
#pragma unroll
    for (int q = 0; q < 4; ++q) acc[mi][q] = vzero8();
  v16bf a0 = load_afrag(A, mbase, kH, 0, ln);
  v16bf a1 = load_afrag(A, mbase + 16, kH, 0, ln);
  v16bf b0 = loadB(0, 0), b1 = loadB(1, 0), b2 = loadB(2, 0), b3 = loadB(3, 0);
  for (int ks = 0; ks < kH / 32 - 1; ++ks) {
    v16bf a0n = load_afrag(A, mbase, kH, (ks + 1) * 32, ln);
    v16bf a1n = load_afrag(A, mbase + 16, kH, (ks + 1) * 32, ln);
    v16bf b0n = loadB(0, ks + 1), b1n = loadB(1, ks + 1);
    v16bf b2n = loadB(2, ks + 1), b3n = loadB(3, ks + 1);
    acc[0][0] = wmma_bf16(a0, b0, acc[0][0]); acc[1][0] = wmma_bf16(a1, b0, acc[1][0]);
    acc[0][1] = wmma_bf16(a0, b1, acc[0][1]); acc[1][1] = wmma_bf16(a1, b1, acc[1][1]);
    acc[0][2] = wmma_bf16(a0, b2, acc[0][2]); acc[1][2] = wmma_bf16(a1, b2, acc[1][2]);
    acc[0][3] = wmma_bf16(a0, b3, acc[0][3]); acc[1][3] = wmma_bf16(a1, b3, acc[1][3]);
    a0 = a0n; a1 = a1n; b0 = b0n; b1 = b1n; b2 = b2n; b3 = b3n;
  }
  acc[0][0] = wmma_bf16(a0, b0, acc[0][0]); acc[1][0] = wmma_bf16(a1, b0, acc[1][0]);
  acc[0][1] = wmma_bf16(a0, b1, acc[0][1]); acc[1][1] = wmma_bf16(a1, b1, acc[1][1]);
  acc[0][2] = wmma_bf16(a0, b2, acc[0][2]); acc[1][2] = wmma_bf16(a1, b2, acc[1][2]);
  acc[0][3] = wmma_bf16(a0, b3, acc[0][3]); acc[1][3] = wmma_bf16(a1, b3, acc[1][3]);
#pragma unroll
  for (int mi = 0; mi < 2; ++mi) {
#pragma unroll
    for (int q = 0; q < 4; ++q) {
      int n = nbase + q * 16 + nl;
      float bo = bout[n];
#pragma unroll
      for (int i = 0; i < 8; ++i) {
        int row = mbase + mi * 16 + i + 8 * (ln >> 4);   // row = t*32 + b
        int t = row >> 5, bb = row & 31;
        out[((size_t)bb * kT + t) * kV + n] = acc[mi][q][i] + bo;
      }
    }
  }
}

// ---------------- launch ----------------
extern "C" void kernel_launch(void* const* d_in, const int* in_sizes, int n_in,
                              void* d_out, int out_size, void* d_ws, size_t ws_size,
                              hipStream_t stream) {
  const int*   tgt   = (const int*)  d_in[0];
  const float* dec   = (const float*)d_in[1];
  const float* cat   = (const float*)d_in[2];
  const float* vfeat = (const float*)d_in[3];
  const float* so    = (const float*)d_in[4];
  const unsigned char* mask = (const unsigned char*)d_in[5];
  const float* emb   = (const float*)d_in[6];
  const float* Wh    = (const float*)d_in[7];
  const float* Ws    = (const float*)d_in[8];
  const float* ab    = (const float*)d_in[9];
  const float* av    = (const float*)d_in[10];
  const float* Wih   = (const float*)d_in[11];
  const float* Whh   = (const float*)d_in[12];
  const float* bih   = (const float*)d_in[13];
  const float* bhh   = (const float*)d_in[14];
  const float* Wout  = (const float*)d_in[15];
  const float* bout  = (const float*)d_in[16];

  float* out  = (float*)d_out;
  float* outH = out + (size_t)kB * kT * kV;

  char* ws = (char*)d_ws;
  unsigned*       bar   = (unsigned*)(ws + OFF_BAR);
  float*          sp    = (float*)(ws + OFF_SP);
  unsigned short* xemb  = (unsigned short*)(ws + OFF_XEMB);
  unsigned short* astep = (unsigned short*)(ws + OFF_ASTEP);
  float*          h2all = (float*)(ws + OFF_H2ALL);
  unsigned short* h2bf  = (unsigned short*)(ws + OFF_H2BF);
  unsigned short* sobf  = (unsigned short*)(ws + OFF_SOBF);
  unsigned short* wst   = (unsigned short*)(ws + OFF_WST);
  unsigned short* wht   = (unsigned short*)(ws + OFF_WHT);
  unsigned short* decbf = (unsigned short*)(ws + OFF_DECBF);
  float*          qbuf  = (float*)(ws + OFF_QBUF);
  unsigned short* wg    = (unsigned short*)(ws + OFF_WG);
  unsigned short* wo    = (unsigned short*)(ws + OFF_WOUT);

  const bool preG = ws_size >= NEED_WG;
  const bool preV = ws_size >= NEED_FULL;

  gd_init<<<1, 128, 0, stream>>>(bar);
  gd_cvt_small<<<544, 64, 0, stream>>>(Ws, Wh, dec, wst, wht, decbf);
  gd_socvt<<<kB * kL, 64, 0, stream>>>(so, sobf);
  if (preG) gd_wgcvt<<<3 * kH, 256, 0, stream>>>(Wih, Whh, wg);
  if (preV) gd_woutcvt<<<(kV * kH) / 2048, 256, 0, stream>>>(Wout, wo);
  gd_sessproj<<<50, 256, 0, stream>>>(sobf, wst, ab, sp);
  gd_xemb<<<kT * kB, 64, 0, stream>>>(tgt, emb, xemb);

  if (preG)
    gd_recur<true><<<kB, 256, 0, stream>>>(dec, cat, vfeat, so, mask, av, Wih, Whh,
                                           bih, bhh, sp, xemb, astep, wg, wht, decbf,
                                           qbuf, h2all, h2bf, outH, bar);
  else
    gd_recur<false><<<kB, 256, 0, stream>>>(dec, cat, vfeat, so, mask, av, Wih, Whh,
                                            bih, bhh, sp, xemb, astep, wg, wht, decbf,
                                            qbuf, h2all, h2bf, outH, bar);

  if (preV)
    gd_vocab<true><<<2000, 256, 0, stream>>>(h2bf, wo, Wout, bout, out);
  else
    gd_vocab<false><<<2000, 256, 0, stream>>>(h2bf, wo, Wout, bout, out);
}
